// SAGE_89429809037918
// MI455X (gfx1250) — matrix operand order repaired and checked by reading, not compiled, not run on hardware
//
#include <hip/hip_runtime.h>
#include <hip/hip_bf16.h>

// ---------------------------------------------------------------------------
// GraphSAGE (pre-Linear + 3x SAGEConv(mean) + L2 normalize) for MI455X gfx1250
//  - GEMMs: v_wmma_f32_16x16x32_bf16 (bf16 in, fp32 accumulate)
//  - A-tile staging: global_load_async_to_lds_b128 (ASYNCcnt), double-buffered
//  - Aggregation: global_atomic_add_f32 on COO edges
// ---------------------------------------------------------------------------

typedef __attribute__((ext_vector_type(16))) __bf16 v16bf;
typedef __attribute__((ext_vector_type(8)))  __bf16 v8bf;
typedef __attribute__((ext_vector_type(8)))  float  v8f;
typedef __attribute__((ext_vector_type(4)))  unsigned int v4u;

__device__ __forceinline__ __bf16 f2bf(float f) {
    unsigned u = __builtin_bit_cast(unsigned, f);
    // round-to-nearest-even bf16
    unsigned r = (u + 0x7FFFu + ((u >> 16) & 1u)) >> 16;
    unsigned short s = (unsigned short)r;
    return __builtin_bit_cast(__bf16, s);
}

// ---------------------------- utility kernels ------------------------------

__global__ void sage_zero_f32(float* __restrict__ p, long n) {
    long i = (long)blockIdx.x * blockDim.x + threadIdx.x;
    if (i < n) p[i] = 0.0f;
}

__global__ void sage_f2b(const float* __restrict__ x, __bf16* __restrict__ y, long n) {
    long i = (long)blockIdx.x * blockDim.x + threadIdx.x;
    if (i < n) y[i] = f2bf(x[i]);
}

__global__ void sage_count(const long long* __restrict__ dst, float* __restrict__ cnt, int E) {
    int e = blockIdx.x * blockDim.x + threadIdx.x;
    if (e < E) atomicAdd(&cnt[(int)dst[e]], 1.0f);
}

// one thread handles 4 consecutive features of one edge
__global__ void sage_scatter(const float* __restrict__ h,
                             const long long* __restrict__ src,
                             const long long* __restrict__ dst,
                             float* __restrict__ agg, int E, int d) {
    long tid = (long)blockIdx.x * blockDim.x + threadIdx.x;
    int per_edge = d >> 2;
    long total = (long)E * per_edge;
    if (tid >= total) return;
    int e  = (int)(tid / per_edge);
    int fb = (int)(tid % per_edge) * 4;
    long s = (long)src[e] * d + fb;
    long t = (long)dst[e] * d + fb;
    float4 v = *(const float4*)(h + s);
    atomicAdd(&agg[t + 0], v.x);
    atomicAdd(&agg[t + 1], v.y);
    atomicAdd(&agg[t + 2], v.z);
    atomicAdd(&agg[t + 3], v.w);
}

// Acat[i] = [ bf16(agg[i]/max(cnt,1)) | bf16(h[i]) ]   (row stride 2d)
__global__ void sage_build_acat(const float* __restrict__ h,
                                const float* __restrict__ agg,
                                const float* __restrict__ cnt,
                                __bf16* __restrict__ Acat, int n, int d) {
    long tid = (long)blockIdx.x * blockDim.x + threadIdx.x;
    if (tid >= (long)n * d) return;
    int i = (int)(tid / d);
    int f = (int)(tid % d);
    float c = cnt[i];
    float mean = agg[tid] / fmaxf(c, 1.0f);
    long base = (long)i * (2 * d);
    Acat[base + f]     = f2bf(mean);
    Acat[base + d + f] = f2bf(h[tid]);
}

// Wcat = [ W_l ; W_r ]  (2d x m, row-major, bf16)
__global__ void sage_build_wcat(const float* __restrict__ Wl,
                                const float* __restrict__ Wr,
                                __bf16* __restrict__ Wcat, int d, int m) {
    long tid = (long)blockIdx.x * blockDim.x + threadIdx.x;
    long total = (long)2 * d * m;
    if (tid >= total) return;
    int row = (int)(tid / m);
    int col = (int)(tid % m);
    float v = (row < d) ? Wl[(long)row * m + col] : Wr[(long)(row - d) * m + col];
    Wcat[tid] = f2bf(v);
}

// ------------------------------ WMMA GEMM ----------------------------------
// out[n x M] = A[n x K](bf16) @ W[K x M](bf16) + bias, optional ReLU.
// Block tile 128x64, 8 waves (4x2), each wave 32x32 via 2x2 wmma 16x16x32.
// A staged with global_load_async_to_lds_b128, double-buffered LDS.

#define BM 128
#define BN 64
#define BK 32
#define LSTRIDE 40   // bf16 units; 80B = 5*16B -> b128-aligned, conflict-free

__global__ __launch_bounds__(256) void sage_gemm_bf16(
    const __bf16* __restrict__ A, const __bf16* __restrict__ W,
    const float* __restrict__ bias, float* __restrict__ out,
    int n, int K, int M, int relu)
{
    __shared__ __align__(16) __bf16 As[2][BM * LSTRIDE];
    __shared__ __align__(16) __bf16 Bs[2][BN * LSTRIDE]; // transposed: [col][k]

    const int tid  = threadIdx.x;
    const int lane = tid & 31;
    const int wave = tid >> 5;
    const int wr   = wave & 3;   // row group (32 rows)
    const int wc   = wave >> 2;  // col group (32 cols)
    const int row0 = blockIdx.x * BM;
    const int col0 = blockIdx.y * BN;

    // ISA 7.12.2 lane decode for 16-bit fragments
    const int frow = lane & 15;
    const int hi   = lane >> 4;        // 0: lanes 0-15, 1: lanes 16-31
    const int a_kb = hi * 8;           // A: K base within 32 (pairs 0..7 / 8..15)
    const int b_kb = hi * 8;           // B: K base within 32 (pairs 0..7 / 8..15)

    // async DMA of the A tile: 512 x 16B chunks, 2 per thread
    // (rows >= n are skipped: their D rows are masked at the epilogue store)
    auto stageA = [&](int buf, int k0) {
        #pragma unroll
        for (int i = 0; i < 2; ++i) {
            int c  = tid + i * 256;
            int r  = c >> 2;          // 0..127
            int co = (c & 3) * 8;     // 0,8,16,24 (bf16)
            int rg = row0 + r;
            if (rg < n) {
                unsigned long long ga =
                    (unsigned long long)(const void*)(A + (long)rg * K + k0 + co);
                unsigned lds = (unsigned)(unsigned long long)(const void*)
                    (&As[buf][r * LSTRIDE + co]);  // flat low32 == LDS offset
                asm volatile("global_load_async_to_lds_b128 %0, %1, off"
                             :: "v"(lds), "v"(ga) : "memory");
            }
        }
    };
    // B tile (32x64) transposed into Bs[buf][col][k] (sync path: needs transpose)
    auto stageB = [&](int buf, int k0) {
        int k = tid >> 3;          // 0..31
        int c = (tid & 7) * 8;     // 0..56
        v4u w  = *(const v4u*)(W + (long)(k0 + k) * M + col0 + c);
        v8bf wv = __builtin_bit_cast(v8bf, w);
        #pragma unroll
        for (int j = 0; j < 8; ++j)
            Bs[buf][(c + j) * LSTRIDE + k] = wv[j];
    };

    v8f acc[2][2] = {};

    int buf = 0;
    stageA(0, 0);
    stageB(0, 0);
    asm volatile("s_wait_asynccnt 0x0" ::: "memory");
    __syncthreads();

    for (int k0 = 0; k0 < K; k0 += BK) {
        // prefetch next K-slab into the other buffer while we compute
        if (k0 + BK < K) {
            stageA(buf ^ 1, k0 + BK);
            stageB(buf ^ 1, k0 + BK);
        }

        v16bf afrag[2], bfrag[2];
        #pragma unroll
        for (int i = 0; i < 2; ++i) {
            int r = wr * 32 + i * 16 + frow;
            const v4u* p = (const v4u*)(&As[buf][r * LSTRIDE + a_kb]);
            union { v16bf v; v4u u[2]; } t;
            t.u[0] = p[0];   // K = kb .. kb+7
            t.u[1] = p[2];   // K = kb+16 .. kb+23  (+32 bytes)
            afrag[i] = t.v;
        }
        #pragma unroll
        for (int j = 0; j < 2; ++j) {
            int cc = wc * 32 + j * 16 + frow;
            const v4u* p = (const v4u*)(&Bs[buf][cc * LSTRIDE + b_kb]);
            union { v16bf v; v4u u[2]; } t;
            t.u[0] = p[0];   // K = kb .. kb+7
            t.u[1] = p[2];   // K = kb+16 .. kb+23  (+32 bytes)
            bfrag[j] = t.v;
        }

        #pragma unroll
        for (int i = 0; i < 2; ++i)
            #pragma unroll
            for (int j = 0; j < 2; ++j)
                acc[i][j] = __builtin_amdgcn_wmma_f32_16x16x32_bf16(
                    false, afrag[i], false, bfrag[j],
                    (short)0, acc[i][j], false, false);

        // own async writes done before the block-wide barrier flips buffers
        asm volatile("s_wait_asynccnt 0x0" ::: "memory");
        __syncthreads();
        buf ^= 1;
    }

    // epilogue: D layout -> M = vgpr + 8*lane[4], N = lane&15
    const int mbase = hi * 8;
    #pragma unroll
    for (int i = 0; i < 2; ++i) {
        #pragma unroll
        for (int j = 0; j < 2; ++j) {
            int ng = col0 + wc * 32 + j * 16 + frow;
            float bv = bias[ng];
            #pragma unroll
            for (int r = 0; r < 8; ++r) {
                int mg = row0 + wr * 32 + i * 16 + mbase + r;
                if (mg < n) {
                    float v = acc[i][j][r] + bv;
                    if (relu) v = fmaxf(v, 0.0f);
                    out[(long)mg * M + ng] = v;
                }
            }
        }
    }
}

// ----------------------------- L2 normalize --------------------------------
// one wave32 per node, 8 nodes per block
__global__ void sage_l2norm(const float* __restrict__ h, float* __restrict__ out,
                            int n, int d) {
    int node = blockIdx.x * 8 + (threadIdx.x >> 5);
    int lane = threadIdx.x & 31;
    if (node >= n) return;
    long base = (long)node * d;
    float s = 0.0f;
    for (int f = lane; f < d; f += 32) {
        float v = h[base + f];
        s += v * v;
    }
    #pragma unroll
    for (int off = 16; off > 0; off >>= 1)
        s += __shfl_xor(s, off, 32);
    float inv = 1.0f / fmaxf(sqrtf(s), 1e-12f);
    for (int f = lane; f < d; f += 32)
        out[base + f] = h[base + f] * inv;
}

// ------------------------------- launcher ----------------------------------

static inline long cdivl(long a, long b) { return (a + b - 1) / b; }

extern "C" void kernel_launch(void* const* d_in, const int* in_sizes, int n_in,
                              void* d_out, int out_size, void* d_ws, size_t ws_size,
                              hipStream_t stream) {
    const int N = in_sizes[0] / 128;
    const int E = in_sizes[1] / 2;

    const float*     x    = (const float*)d_in[0];
    const long long* ei   = (const long long*)d_in[1];   // int64 (2,E)
    const long long* srcI = ei;
    const long long* dstI = ei + E;
    const float* preW = (const float*)d_in[2];
    const float* preb = (const float*)d_in[3];
    const float* W1l  = (const float*)d_in[4];
    const float* b1   = (const float*)d_in[5];
    const float* W1r  = (const float*)d_in[6];
    const float* W2l  = (const float*)d_in[7];
    const float* b2   = (const float*)d_in[8];
    const float* W2r  = (const float*)d_in[9];
    const float* W3l  = (const float*)d_in[10];
    const float* b3   = (const float*)d_in[11];
    const float* W3r  = (const float*)d_in[12];

    // workspace layout (256B aligned slices)
    char* ws = (char*)d_ws;
    size_t off = 0;
    auto alloc = [&](size_t bytes) -> void* {
        void* p = ws + off;
        off += (bytes + 255) & ~(size_t)255;
        return p;
    };
    float*  hA   = (float*)alloc((size_t)N * 256 * sizeof(float));
    float*  hB   = (float*)alloc((size_t)N * 256 * sizeof(float));
    float*  agg  = (float*)alloc((size_t)N * 256 * sizeof(float));
    float*  cnt  = (float*)alloc((size_t)N * sizeof(float));
    __bf16* Acat = (__bf16*)alloc((size_t)N * 512 * sizeof(__bf16));
    __bf16* Wcat = (__bf16*)alloc((size_t)512 * 256 * sizeof(__bf16));
    (void)ws_size; (void)n_in;

    const int T = 256;

    // in-degree counts (same for all layers)
    sage_zero_f32<<<cdivl(N, T), T, 0, stream>>>(cnt, N);
    sage_count<<<cdivl(E, T), T, 0, stream>>>(dstI, cnt, E);

    // pre: h0 = x @ preW + preb   (K=128, M=128)
    sage_f2b<<<cdivl((long)N * 128, T), T, 0, stream>>>(x, Acat, (long)N * 128);
    sage_f2b<<<cdivl(128L * 128, T), T, 0, stream>>>(preW, Wcat, 128L * 128);
    sage_gemm_bf16<<<dim3((unsigned)cdivl(N, BM), 128 / BN), T, 0, stream>>>(
        Acat, Wcat, preb, hA, N, 128, 128, 0);

    // one SAGE layer: hout = relu?([mean|hin] @ [Wl;Wr] + b)
    auto run_layer = [&](const float* hin, int din,
                         const float* Wl, const float* b, const float* Wr,
                         float* hout, int relu) {
        long aggN = (long)N * din;
        sage_zero_f32<<<cdivl(aggN, T), T, 0, stream>>>(agg, aggN);
        sage_scatter<<<cdivl((long)E * (din / 4), T), T, 0, stream>>>(
            hin, srcI, dstI, agg, E, din);
        sage_build_acat<<<cdivl((long)N * din, T), T, 0, stream>>>(
            hin, agg, cnt, Acat, N, din);
        sage_build_wcat<<<cdivl((long)2 * din * 256, T), T, 0, stream>>>(
            Wl, Wr, Wcat, din, 256);
        sage_gemm_bf16<<<dim3((unsigned)cdivl(N, BM), 256 / BN), T, 0, stream>>>(
            Acat, Wcat, b, hout, N, 2 * din, 256, relu);
    };

    run_layer(hA, 128, W1l, b1, W1r, hB, 1);   // conv_first + relu
    run_layer(hB, 256, W2l, b2, W2r, hA, 1);   // conv_hidden + relu
    run_layer(hA, 256, W3l, b3, W3r, hB, 0);   // conv_out

    sage_l2norm<<<cdivl(N, 8), T, 0, stream>>>(hB, (float*)d_out, N, 256);
    (void)out_size;
}
